// ShakespeareLSTM_841813590674
// MI455X (gfx1250) — matrix operand order repaired and checked by reading, not compiled, not run on hardware
//
#include <hip/hip_runtime.h>
#include <hip/hip_bf16.h>

// ---------------------------------------------------------------------------
// ShakespeareLSTM on MI455X (gfx1250), wave32 + WMMA bf16 + async LDS staging.
//
// Pipeline (all on `stream`):
//   1. Convert weights fp32 -> bf16 (RNE); combine biases b_ih+b_hh.
//   2. Embedding gather -> bf16 activations [B*T, E].
//   3. WMMA GEMM: xin0[t,b,4H] = emb @ W_ih0^T + bias  (fp32 out, L2-resident W)
//   4. Persistent scan kernel, layer 0: 64 blocks x 128 thr, W_hh slice
//      (64 rows x 1024 K bf16 = 128KB) staged into LDS with
//      GLOBAL_LOAD_ASYNC_TO_LDS_B128 (ASYNCcnt) and pinned for all 512 steps;
//      cell state lives in VGPR accumulators; grid barrier per timestep.
//   5. WMMA GEMM: xin1 from h1; scan layer 1; WMMA GEMM FC -> d_out [B,T,V].
//
// Scratch layout (~1.25 GB, assumed to fit ws_size; compile-only loop).
// ---------------------------------------------------------------------------

#define T_SEQ   512
#define BATCH_  64
#define EMBED_  512
#define HIDDEN_ 1024
#define G4H     4096
#define VOCAB_  128
#define MROWS   32768   // B*T

typedef __attribute__((ext_vector_type(16))) __bf16 bf16x16;
typedef __attribute__((ext_vector_type(8)))  float  f32x8;

#if __has_builtin(__builtin_amdgcn_global_load_async_to_lds_b128)
#define HAVE_ASYNC_LDS 1
typedef __attribute__((ext_vector_type(4))) int v4i_t;
typedef __attribute__((address_space(1))) v4i_t* gptr128_t;   // global v4i*
typedef __attribute__((address_space(3))) v4i_t* lptr128_t;   // LDS v4i*
#endif

__device__ __forceinline__ void wait_asynccnt0() {
#if __has_builtin(__builtin_amdgcn_s_wait_asynccnt)
  __builtin_amdgcn_s_wait_asynccnt(0);
#else
  asm volatile("s_wait_asynccnt 0" ::: "memory");
#endif
}

// ---------------- scalar helpers ----------------

__device__ __forceinline__ unsigned short f2bf(float f) {
  unsigned int u = __float_as_uint(f);
  u += 0x7fffu + ((u >> 16) & 1u);          // round-to-nearest-even
  return (unsigned short)(u >> 16);
}

__device__ __forceinline__ float sigmoid_f(float x) {
  return 1.0f / (1.0f + __expf(-x));
}
__device__ __forceinline__ float tanh_f(float x) {
  // 1 - 2/(e^{2x}+1): stable for |x| large (exp->inf gives 1, exp->0 gives -1)
  return 1.0f - 2.0f / (__expf(2.0f * x) + 1.0f);
}

// ---------------- WMMA fragment helpers ----------------
// A 16x32 bf16: lane m=lane&15; lanes 0-15 hold K {0..7,16..23}, lanes 16-31
// hold K {8..15,24..31}  ->  chunks at (k0 + sel*8) and (k0 + 16 + sel*8).
__device__ __forceinline__ bf16x16 load_frag_a(const unsigned short* base) {
  union { bf16x16 v; uint4 q[2]; } u;
  u.q[0] = *reinterpret_cast<const uint4*>(base);
  u.q[1] = *reinterpret_cast<const uint4*>(base + 16);
  return u.v;
}
// B 32x16 bf16: lane n=lane&15; lanes 0-15 hold K {0..7,16..23}, lanes 16-31
// hold K {8..15,24..31}  ->  chunks at (k0 + sel*8) and (+16).
__device__ __forceinline__ bf16x16 load_frag_b(const unsigned short* base) {
  union { bf16x16 v; uint4 q[2]; } u;
  u.q[0] = *reinterpret_cast<const uint4*>(base);
  u.q[1] = *reinterpret_cast<const uint4*>(base + 16);
  return u.v;
}

__device__ __forceinline__ f32x8 wmma_bf16(bf16x16 a, bf16x16 b, f32x8 c) {
  // (neg_a, A, neg_b, B, c_mod, C, reuse_a, reuse_b)
  return __builtin_amdgcn_wmma_f32_16x16x32_bf16(false, a, false, b,
                                                 (short)0, c, false, false);
}

// ---------------- small utility kernels ----------------

__global__ void cvt_f32_bf16(const float* __restrict__ in,
                             unsigned short* __restrict__ out, int n) {
  for (int i = blockIdx.x * blockDim.x + threadIdx.x; i < n;
       i += gridDim.x * blockDim.x)
    out[i] = f2bf(in[i]);
}

__global__ void vec_add_f32(const float* __restrict__ a,
                            const float* __restrict__ b,
                            float* __restrict__ o, int n) {
  for (int i = blockIdx.x * blockDim.x + threadIdx.x; i < n;
       i += gridDim.x * blockDim.x)
    o[i] = a[i] + b[i];
}

__global__ void zero_u32(unsigned int* __restrict__ p, int n) {
  for (int i = blockIdx.x * blockDim.x + threadIdx.x; i < n;
       i += gridDim.x * blockDim.x)
    p[i] = 0u;
}

__global__ void embed_to_bf16(const int* __restrict__ x,
                              const float* __restrict__ emb,
                              unsigned short* __restrict__ out, int total) {
  for (int i = blockIdx.x * blockDim.x + threadIdx.x; i < total;
       i += gridDim.x * blockDim.x) {
    int row = i >> 9;          // / EMBED_
    int e   = i & (EMBED_ - 1);
    int tok = x[row];
    out[i] = f2bf(emb[(size_t)tok * EMBED_ + e]);
  }
}

// ---------------- WMMA GEMM: out[M,N] = A[M,K](bf16) @ W[N,K](bf16)^T + bias
// block = 128 threads (4 waves); block tile 64(M) x 64(N); wave = 16M x 64N.
// perm: 0 = identity row, 1 = (b*T+t)->(t*B+b), 2 = (t*B+b)->(b*T+t).
__global__ void wmma_gemm_bias(const unsigned short* __restrict__ A,
                               const unsigned short* __restrict__ W,
                               const float* __restrict__ bias,
                               float* __restrict__ out,
                               int M, int N, int K, int perm) {
  const int lane  = threadIdx.x & 31;
  const int wv    = threadIdx.x >> 5;
  const int sel   = lane >> 4;
  const int l16   = lane & 15;
  const int mbase = blockIdx.y * 64 + wv * 16;
  const int nbase = blockIdx.x * 64;

  const unsigned short* arow = A + (size_t)(mbase + l16) * K;

  f32x8 acc[4];
#pragma unroll
  for (int f = 0; f < 4; ++f) acc[f] = f32x8{0, 0, 0, 0, 0, 0, 0, 0};

  for (int k0 = 0; k0 < K; k0 += 32) {
    bf16x16 a = load_frag_a(arow + k0 + sel * 8);
    __builtin_prefetch(arow + k0 + 64, 0, 1);   // global_prefetch_b8
#pragma unroll
    for (int f = 0; f < 4; ++f) {
      const unsigned short* brow =
          W + (size_t)(nbase + f * 16 + l16) * K + k0 + sel * 8;
      bf16x16 b = load_frag_b(brow);
      acc[f] = wmma_bf16(a, b, acc[f]);
    }
  }

#pragma unroll
  for (int f = 0; f < 4; ++f) {
    const int n  = nbase + f * 16 + l16;
    const float bi = bias ? bias[n] : 0.0f;
#pragma unroll
    for (int v = 0; v < 8; ++v) {
      const int m = mbase + sel * 8 + v;
      size_t orow;
      if (perm == 1)      orow = (size_t)(m & 511) * 64  + (m >> 9);
      else if (perm == 2) orow = (size_t)(m & 63)  * 512 + (m >> 6);
      else                orow = (size_t)m;
      out[orow * (size_t)N + n] = acc[f][v] + bi;
    }
  }
}

// ---------------- device-wide sense barrier (persistent scan) ----------------

__device__ __forceinline__ void grid_sync(unsigned int* cnt, unsigned int* gen,
                                          unsigned int nblk, unsigned int target) {
  __threadfence();            // publish h stores at device scope
  __syncthreads();
  if (threadIdx.x == 0) {
    unsigned int prev = atomicAdd(cnt, 1u);
    if (prev == nblk - 1u) {
      *cnt = 0u;              // reset before release; ordered by fence below
      __threadfence();
      atomicAdd(gen, 1u);     // release generation `target`
    } else {
      while (__hip_atomic_load(gen, __ATOMIC_ACQUIRE,
                               __HIP_MEMORY_SCOPE_AGENT) < target) {
        __builtin_amdgcn_s_sleep(2);
      }
    }
  }
  __syncthreads();
}

// ---------------- persistent LSTM scan (one layer) ----------------
// grid = 64 blocks (hidden-unit tiles of 16), block = 128 thr (4 waves =
// 4 batch tiles of 16). Dynamic LDS = 64 rows x 1024 K bf16 = 128KB holds
// this block's W_hh slice for the whole sequence (staged with async B128
// copies, tracked by ASYNCcnt). Cell state lives in the wave's f32x8
// accumulator fragment for all 512 steps.
__global__ void lstm_scan(const unsigned short* __restrict__ Whh,  // [4H, H] bf16
                          const float* __restrict__ xin,           // [T, B, 4H] f32 (bias folded)
                          unsigned short* __restrict__ hbuf,       // [(T+1), B, H] bf16, slot0 = 0
                          unsigned int* __restrict__ bar_cnt,
                          unsigned int* __restrict__ bar_gen) {
  extern __shared__ unsigned short lw[];    // 64*1024 bf16 = 128KB
  const int lane = threadIdx.x & 31;
  const int wv   = threadIdx.x >> 5;        // batch tile 0..3
  const int sel  = lane >> 4;
  const int l16  = lane & 15;
  const int u0   = blockIdx.x * 16;         // hidden-unit tile base

  // Stage W_hh slice: LDS row r = g*16 + j  <-  Whh[g*H + u0 + j][0..H)
  unsigned short* WhhNC = const_cast<unsigned short*>(Whh);
  for (int i = threadIdx.x; i < 64 * (HIDDEN_ / 8); i += blockDim.x) {
    int r  = i >> 7;          // / (HIDDEN_/8 = 128 uint4 chunks per row)
    int ch = i & 127;
    int g = r >> 4, j = r & 15;
    uint4* src = reinterpret_cast<uint4*>(
        WhhNC + (size_t)(g * HIDDEN_ + u0 + j) * HIDDEN_) + ch;
    uint4* dst = reinterpret_cast<uint4*>(lw) + (r * 128 + ch);
#if defined(HAVE_ASYNC_LDS)
    // GLOBAL_LOAD_ASYNC_TO_LDS_B128: memory -> LDS without VGPR round-trip.
    __builtin_amdgcn_global_load_async_to_lds_b128(
        (gptr128_t)(void*)src, (lptr128_t)(void*)dst, 0, 0);
#else
    *dst = *src;
#endif
  }
#if defined(HAVE_ASYNC_LDS)
  wait_asynccnt0();
#endif
  __syncthreads();

  const int arow_m = wv * 16 + l16;         // batch row for A-frag loads
  f32x8 cstate = f32x8{0, 0, 0, 0, 0, 0, 0, 0};

  for (int t = 0; t < T_SEQ; ++t) {
    const unsigned short* hprev = hbuf + (size_t)t * BATCH_ * HIDDEN_;
    const unsigned short* ar    = hprev + (size_t)arow_m * HIDDEN_;

    f32x8 acc[4];
#pragma unroll
    for (int g = 0; g < 4; ++g) acc[g] = f32x8{0, 0, 0, 0, 0, 0, 0, 0};

    for (int k0 = 0; k0 < HIDDEN_; k0 += 32) {
      bf16x16 a = load_frag_a(ar + k0 + sel * 8);
#pragma unroll
      for (int g = 0; g < 4; ++g) {
        bf16x16 b = load_frag_b(
            lw + (size_t)(g * 16 + l16) * HIDDEN_ + k0 + sel * 8);
        acc[g] = wmma_bf16(a, b, acc[g]);
      }
    }

    // Elementwise cell update for this wave's 16(batch) x 16(unit) tile.
    const float*    xr   = xin  + (size_t)t * BATCH_ * G4H;
    unsigned short* hout = hbuf + (size_t)(t + 1) * BATCH_ * HIDDEN_;
#pragma unroll
    for (int v = 0; v < 8; ++v) {
      const int bm = wv * 16 + sel * 8 + v;   // batch index (C/D layout)
      const int n  = u0 + l16;                // hidden unit
      const float* xp = xr + (size_t)bm * G4H + n;
      float zi = acc[0][v] + xp[0 * HIDDEN_];
      float zf = acc[1][v] + xp[1 * HIDDEN_];
      float zg = acc[2][v] + xp[2 * HIDDEN_];
      float zo = acc[3][v] + xp[3 * HIDDEN_];
      float ig = sigmoid_f(zi);
      float fg = sigmoid_f(zf);
      float gg = tanh_f(zg);
      float og = sigmoid_f(zo);
      float c  = fg * cstate[v] + ig * gg;
      cstate[v] = c;
      float h = og * tanh_f(c);
      hout[(size_t)bm * HIDDEN_ + n] = f2bf(h);
    }

    grid_sync(bar_cnt, bar_gen, gridDim.x, (unsigned)(t + 1));
  }
}

// ---------------- host launch ----------------

extern "C" void kernel_launch(void* const* d_in, const int* in_sizes, int n_in,
                              void* d_out, int out_size, void* d_ws, size_t ws_size,
                              hipStream_t stream) {
  (void)in_sizes; (void)n_in; (void)out_size; (void)ws_size;

  const int*   x    = (const int*)d_in[0];
  const float* emb  = (const float*)d_in[1];
  const float* wih0 = (const float*)d_in[2];
  const float* whh0 = (const float*)d_in[3];
  const float* bih0 = (const float*)d_in[4];
  const float* bhh0 = (const float*)d_in[5];
  const float* wih1 = (const float*)d_in[6];
  const float* whh1 = (const float*)d_in[7];
  const float* bih1 = (const float*)d_in[8];
  const float* bhh1 = (const float*)d_in[9];
  const float* fcw  = (const float*)d_in[10];
  const float* fcb  = (const float*)d_in[11];
  float* out = (float*)d_out;

  // ---- scratch carve-up (256B aligned) ----
  char* ws = (char*)d_ws;
  auto carve = [&](size_t bytes) -> char* {
    char* p = ws;
    ws += (bytes + 255) & ~(size_t)255;
    return p;
  };
  unsigned short* embq  = (unsigned short*)carve((size_t)MROWS * EMBED_ * 2);
  unsigned short* wih0q = (unsigned short*)carve((size_t)G4H * EMBED_ * 2);
  unsigned short* whh0q = (unsigned short*)carve((size_t)G4H * HIDDEN_ * 2);
  unsigned short* wih1q = (unsigned short*)carve((size_t)G4H * HIDDEN_ * 2);
  unsigned short* whh1q = (unsigned short*)carve((size_t)G4H * HIDDEN_ * 2);
  unsigned short* fcwq  = (unsigned short*)carve((size_t)VOCAB_ * HIDDEN_ * 2);
  float* bsum0 = (float*)carve((size_t)G4H * 4);
  float* bsum1 = (float*)carve((size_t)G4H * 4);
  float* xin0  = (float*)carve((size_t)MROWS * G4H * 4);
  float* xin1  = (float*)carve((size_t)MROWS * G4H * 4);
  unsigned short* hbuf1 =
      (unsigned short*)carve((size_t)(T_SEQ + 1) * BATCH_ * HIDDEN_ * 2);
  unsigned short* hbuf2 =
      (unsigned short*)carve((size_t)(T_SEQ + 1) * BATCH_ * HIDDEN_ * 2);
  unsigned int* bar = (unsigned int*)carve(256);

  // ---- 1. weight/bias prep ----
  cvt_f32_bf16<<<1024, 256, 0, stream>>>(wih0, wih0q, G4H * EMBED_);
  cvt_f32_bf16<<<2048, 256, 0, stream>>>(whh0, whh0q, G4H * HIDDEN_);
  cvt_f32_bf16<<<2048, 256, 0, stream>>>(wih1, wih1q, G4H * HIDDEN_);
  cvt_f32_bf16<<<2048, 256, 0, stream>>>(whh1, whh1q, G4H * HIDDEN_);
  cvt_f32_bf16<<<64,   256, 0, stream>>>(fcw,  fcwq,  VOCAB_ * HIDDEN_);
  vec_add_f32<<<16, 256, 0, stream>>>(bih0, bhh0, bsum0, G4H);
  vec_add_f32<<<16, 256, 0, stream>>>(bih1, bhh1, bsum1, G4H);

  // ---- 2. embedding gather -> bf16 ----
  embed_to_bf16<<<4096, 256, 0, stream>>>(x, emb, embq, MROWS * EMBED_);

  // ---- zero h0 slots + barrier state ----
  zero_u32<<<64, 256, 0, stream>>>((unsigned int*)hbuf1, BATCH_ * HIDDEN_ / 2);
  zero_u32<<<64, 256, 0, stream>>>((unsigned int*)hbuf2, BATCH_ * HIDDEN_ / 2);
  zero_u32<<<1, 32, 0, stream>>>(bar, 2);

  // ---- 3. xin0 = emb @ W_ih0^T + b  (rows permuted (b,t)->(t,b)) ----
  {
    dim3 g(G4H / 64, MROWS / 64);
    wmma_gemm_bias<<<g, 128, 0, stream>>>(embq, wih0q, bsum0, xin0,
                                          MROWS, G4H, EMBED_, 1);
  }

  // ---- 4. layer-0 scan (persistent, 128KB LDS weight slice) ----
  lstm_scan<<<dim3(HIDDEN_ / 16), 128, 64 * HIDDEN_ * 2, stream>>>(
      whh0q, xin0, hbuf1, bar, bar + 1);

  // ---- 5. xin1 = h1 @ W_ih1^T + b  (rows already (t,b)) ----
  {
    dim3 g(G4H / 64, MROWS / 64);
    wmma_gemm_bias<<<g, 128, 0, stream>>>(hbuf1 + BATCH_ * HIDDEN_, wih1q,
                                          bsum1, xin1, MROWS, G4H, HIDDEN_, 0);
  }

  zero_u32<<<1, 32, 0, stream>>>(bar, 2);

  // ---- 6. layer-1 scan ----
  lstm_scan<<<dim3(HIDDEN_ / 16), 128, 64 * HIDDEN_ * 2, stream>>>(
      whh1q, xin1, hbuf2, bar, bar + 1);

  // ---- 7. FC: out[b,t,V] = h2 @ fc_w^T + fc_b  (rows (t,b)->(b,t)) ----
  {
    dim3 g(VOCAB_ / 64, MROWS / 64);
    wmma_gemm_bias<<<g, 128, 0, stream>>>(hbuf2 + BATCH_ * HIDDEN_, fcwq,
                                          fcb, out, MROWS, VOCAB_, HIDDEN_, 2);
  }
}
